// ANFIS_61280593380167
// MI455X (gfx1250) — matrix operand order repaired and checked by reading, not compiled, not run on hardware
//
#include <hip/hip_runtime.h>
#include <hip/hip_bf16.h>

typedef _Float16 v16h __attribute__((ext_vector_type(16)));
typedef _Float16 v8h  __attribute__((ext_vector_type(8)));
typedef float    v8f  __attribute__((ext_vector_type(8)));

#define B_TOT 8192
#define D_TOT 256
#define R_TOT 64
#define O_TOT 256
#define KC_TOTAL 514          // 512 data K-chunks of 32 + 2 bias chunks
#define NT_TOT (O_TOT / 16)   // 16 N-tiles

__device__ __forceinline__ v8f wmma16(v16h a, v16h b, v8f c) {
  // (neg_a, A, neg_b, B, c_mod, C, reuse_a, reuse_b)
  return __builtin_amdgcn_wmma_f32_16x16x32_f16(false, a, false, b, (short)0, c,
                                                false, false);
}

__device__ __forceinline__ v16h splat16(_Float16 s) {
  v16h v;
#pragma unroll
  for (int e = 0; e < 16; ++e) v[e] = s;
  return v;
}

// WGP-scope prefetch (no scope modifier == scope 0: pull into all cache
// levels incl. the WGP cache; the __builtin_prefetch path only reaches
// SE/SYS scope which stops at GL2 where W already resides).
__device__ __forceinline__ void prefetch_wgp(const void* p) {
  asm volatile("global_prefetch_b8 %0, off" ::"v"(p));
}

// ---------------------------------------------------------------------------
// Kernel 0: precompute -1/(2*sig^2) so the membership loop is a pure FMA loop
// (avoids 8192x replicating the exact-division sequence).
// ---------------------------------------------------------------------------
__global__ void prep_ms_kernel(const float* __restrict__ sig,
                               float* __restrict__ neginv) {
  const int i = blockIdx.x * 256 + threadIdx.x;  // 16384 elements
  const float s = sig[i];
  neginv[i] = -1.0f / (2.0f * s * s);
}

// ---------------------------------------------------------------------------
// Kernel 1: Gaussian memberships + normalization.  Accumulated in f32
// (matches the reference's f32 exp underflow semantics); the normalized
// weights are emitted transposed in f16, fT[r][b], so the GEMM kernel can
// fetch per-rule weights with coalesced u16 loads and no converts.
// ---------------------------------------------------------------------------
__global__ void memberships_kernel(const float* __restrict__ x,
                                   const float* __restrict__ mu,
                                   const float* __restrict__ neginv,
                                   _Float16* __restrict__ fT) {
  __shared__ float xs[D_TOT];
  __shared__ float red[R_TOT];
  const int b = blockIdx.x;
  const int r = threadIdx.x;  // 64 threads = 2 waves
#pragma unroll
  for (int j = 0; j < D_TOT / R_TOT; ++j)
    xs[r + j * R_TOT] = x[(size_t)b * D_TOT + r + j * R_TOT];
  __syncthreads();
  float acc = 0.f;
  for (int i = 0; i < D_TOT; ++i) {
    const float m = mu[i * R_TOT + r];      // [D,R] -> coalesced over r
    const float ni = neginv[i * R_TOT + r]; // -1/(2 sig^2)
    const float d = xs[i] - m;
    acc = fmaf(d * d, ni, acc);
  }
  const float frs = expf(acc);  // AMPLI == 0
  red[r] = frs;
  __syncthreads();
  float sum = 0.f;
#pragma unroll
  for (int j = 0; j < R_TOT; ++j) sum += red[j];  // broadcast LDS reads
  fT[(size_t)r * B_TOT + b] = (_Float16)(frs / (sum + 1e-8f));
}

// ---------------------------------------------------------------------------
// Kernel 2: swizzle W (f32 [16384,256]) + bias (f32 [64,256]) into f16 WMMA
// B-fragment layout: block (kc,nt) holds a 32x16 (KxN) tile, lane L owns
// column n = nt*16+(L&15), k = kc*32 + (e&8)*2 + (L>>4)*8 + (e&7)   (e = 0..15).
// ---------------------------------------------------------------------------
__global__ void prep_w_kernel(const float* __restrict__ W,
                              const float* __restrict__ bias,
                              _Float16* __restrict__ Whf) {
  const int unit = blockIdx.x * 4 + (threadIdx.x >> 5);  // kc*16 + nt
  const int lane = threadIdx.x & 31;
  if (unit >= KC_TOTAL * NT_TOT) return;
  const int kc = unit >> 4, nt = unit & 15;
  const int half = lane >> 4, l15 = lane & 15;
  const int n = nt * 16 + l15;
  v16h v;
#pragma unroll
  for (int e = 0; e < 16; ++e) {
    const int k = kc * 32 + ((e & 8) << 1) + half * 8 + (e & 7);
    const float s = (k < 16384) ? W[(size_t)k * O_TOT + n]
                                : bias[(size_t)(k - 16384) * O_TOT + n];
    v[e] = (_Float16)s;
  }
  *(v16h*)(Whf + ((size_t)unit * 32 + lane) * 16) = v;
}

// ---------------------------------------------------------------------------
// Kernel 3: swizzle x (f32 [8192,256]) into f16 WMMA A-fragment layout:
// block (mt,ic) holds a 16x32 (MxK) tile, lane L owns row m = mt*16+(L&15),
// kk = (e&8)*2 + (L>>4)*8 + (e&7).
// ---------------------------------------------------------------------------
__global__ void prep_x_kernel(const float* __restrict__ x,
                              _Float16* __restrict__ Xf) {
  const int unit = blockIdx.x * 4 + (threadIdx.x >> 5);  // mt*8 + ic
  const int lane = threadIdx.x & 31;
  if (unit >= (B_TOT / 16) * 8) return;
  const int mt = unit >> 3, ic = unit & 7;
  const int half = lane >> 4, l15 = lane & 15;
  const int m = mt * 16 + l15;
  v16h v;
#pragma unroll
  for (int e = 0; e < 16; ++e) {
    const int kk = ((e & 8) << 1) + half * 8 + (e & 7);
    v[e] = (_Float16)x[(size_t)m * D_TOT + ic * 32 + kk];
  }
  *(v16h*)(Xf + ((size_t)unit * 32 + lane) * 16) = v;
}

// ---------------------------------------------------------------------------
// Kernel 4: out = [f*x | f] @ [W ; bias]  via f16 WMMA, f32 accumulate.
// Workgroup = 4 waves (2x2), block tile 64(M) x 64(N); wave tile 32x32
// (2 M-tiles x 2 N-tiles) -> 512 blocks / 2048 waves for occupancy.
// No LDS, no barriers: fragments stream from the pre-swizzled buffers with
// fully coalesced b128 loads (W = 8.4 MB f16, resident in the 192 MB L2),
// with a WGP-scope prefetch running 2 K-chunks ahead, alternating between
// the wave's two B streams so both are covered.
// ---------------------------------------------------------------------------
__global__ __launch_bounds__(128) void anfis_gemm_kernel(
    const _Float16* __restrict__ fT, const _Float16* __restrict__ Whf,
    const _Float16* __restrict__ Xf, float* __restrict__ out) {
  const int lane = threadIdx.x & 31;
  const int w = threadIdx.x >> 5;
  const int wm = w >> 1, wn = w & 1;
  const int half = lane >> 4, l15 = lane & 15;
  const int m0 = blockIdx.x * 64 + wm * 32;
  const int n0 = blockIdx.y * 64 + wn * 32;
  const int mt0 = m0 >> 4, nt0 = n0 >> 4;

  const v8h* __restrict__ Xb = (const v8h*)Xf;
  const v8h* __restrict__ Wb = (const v8h*)Whf;

  v8f acc[2][2] = {};

  for (int r = 0; r < R_TOT; ++r) {
    // per-lane rule weight for each M-tile: coalesced u16 loads (lanes 0-15
    // cover 32 contiguous bytes of fT[r][...])
    const _Float16 fh0 = fT[(size_t)r * B_TOT + m0 + l15];
    const _Float16 fh1 = fT[(size_t)r * B_TOT + m0 + 16 + l15];
    const v16h fs0 = splat16(fh0);
    const v16h fs1 = splat16(fh1);
#pragma unroll
    for (int ic = 0; ic < 8; ++ic) {
      const int kc = r * 8 + ic;
      // A fragments: x fragment * f[b,r]  (v_pk_mul_f16); the Xf loads are
      // r-invariant so the compiler hoists/caches them (L0-resident anyway).
      const v8h* p0 = Xb + ((size_t)((mt0 + 0) * 8 + ic) * 32 + lane) * 2;
      const v8h* p1 = Xb + ((size_t)((mt0 + 1) * 8 + ic) * 32 + lane) * 2;
      v16h a0 = __builtin_shufflevector(p0[0], p0[1], 0, 1, 2, 3, 4, 5, 6, 7, 8,
                                        9, 10, 11, 12, 13, 14, 15) * fs0;
      v16h a1 = __builtin_shufflevector(p1[0], p1[1], 0, 1, 2, 3, 4, 5, 6, 7, 8,
                                        9, 10, 11, 12, 13, 14, 15) * fs1;
      // B fragments (coalesced: lane-consecutive 32B chunks)
      v16h bb[2];
#pragma unroll
      for (int u = 0; u < 2; ++u) {
        const v8h* q = Wb + ((size_t)(kc * 16 + nt0 + u) * 32 + lane) * 2;
        bb[u] = __builtin_shufflevector(q[0], q[1], 0, 1, 2, 3, 4, 5, 6, 7, 8,
                                        9, 10, 11, 12, 13, 14, 15);
      }
      // WGP-scope prefetch of W two K-chunks ahead; one 1KB B block per
      // prefetch, alternating nt so both of this wave's streams are covered.
      prefetch_wgp((const void*)(
          Wb + ((size_t)((kc + 2) * 16 + nt0 + (ic & 1)) * 32 + lane) * 2));
#pragma unroll
      for (int u = 0; u < 2; ++u) acc[0][u] = wmma16(a0, bb[u], acc[0][u]);
#pragma unroll
      for (int u = 0; u < 2; ++u) acc[1][u] = wmma16(a1, bb[u], acc[1][u]);
    }
  }

  // Bias tail: 2 extra K-chunks where A[m, 16384+r] = f[m,r]
#pragma unroll
  for (int tail = 0; tail < 2; ++tail) {
    const int kc = 512 + tail;
    v16h a[2];
#pragma unroll
    for (int t = 0; t < 2; ++t) {
#pragma unroll
      for (int e = 0; e < 16; ++e) {
        const int kk = ((e & 8) << 1) + half * 8 + (e & 7);
        a[t][e] = fT[(size_t)(tail * 32 + kk) * B_TOT + m0 + t * 16 + l15];
      }
    }
    v16h bb[2];
#pragma unroll
    for (int u = 0; u < 2; ++u) {
      const v8h* q = Wb + ((size_t)(kc * 16 + nt0 + u) * 32 + lane) * 2;
      bb[u] = __builtin_shufflevector(q[0], q[1], 0, 1, 2, 3, 4, 5, 6, 7, 8, 9,
                                      10, 11, 12, 13, 14, 15);
    }
#pragma unroll
    for (int t = 0; t < 2; ++t)
#pragma unroll
      for (int u = 0; u < 2; ++u) acc[t][u] = wmma16(a[t], bb[u], acc[t][u]);
  }

  // Store: C/D layout -> lane gives N, (half,vgpr) gives M. Coalesced over l15.
#pragma unroll
  for (int t = 0; t < 2; ++t)
#pragma unroll
    for (int u = 0; u < 2; ++u) {
      const int n = n0 + u * 16 + l15;
#pragma unroll
      for (int v = 0; v < 8; ++v) {
        const int m = m0 + t * 16 + half * 8 + v;
        out[(size_t)m * O_TOT + n] = acc[t][u][v];
      }
    }
}

// ---------------------------------------------------------------------------
extern "C" void kernel_launch(void* const* d_in, const int* in_sizes, int n_in,
                              void* d_out, int out_size, void* d_ws,
                              size_t ws_size, hipStream_t stream) {
  const float* x = (const float*)d_in[0];     // [8192,256]
  const float* mu = (const float*)d_in[1];    // [256,64]
  const float* sig = (const float*)d_in[2];   // [256,64]
  const float* W = (const float*)d_in[3];     // [64,256,256]
  const float* bias = (const float*)d_in[4];  // [64,256]
  float* out = (float*)d_out;                 // [8192,256]

  // workspace layout
  _Float16* fT = (_Float16*)d_ws;  // 64 x 8192 f16 = 1 MB (transposed f)
  _Float16* Whf = (_Float16*)((char*)d_ws + (size_t)1024 * 1024);
  const size_t whf_bytes = (size_t)KC_TOTAL * 16 * 32 * 16 * 2;  // 8.42 MB
  _Float16* Xf = (_Float16*)((char*)Whf + whf_bytes);            // 4 MB
  float* neginv = (float*)((char*)Xf + (size_t)B_TOT * D_TOT * 2);  // 64 KB

  prep_ms_kernel<<<(D_TOT * R_TOT) / 256, 256, 0, stream>>>(sig, neginv);
  memberships_kernel<<<B_TOT, R_TOT, 0, stream>>>(x, mu, neginv, fT);
  prep_w_kernel<<<(KC_TOTAL * NT_TOT) / 4, 128, 0, stream>>>(W, bias, Whf);
  prep_x_kernel<<<((B_TOT / 16) * 8) / 4, 128, 0, stream>>>(x, Xf);
  anfis_gemm_kernel<<<dim3(B_TOT / 64, O_TOT / 64), 128, 0, stream>>>(
      fT, Whf, Xf, out);
}
